// DSDLoss_10514079941330
// MI455X (gfx1250) — compile-verified, matrix operand order checked by reading, not run on hardware
//
#include <hip/hip_runtime.h>
#include <math.h>

// Problem constants (from reference)
#define BB    8
#define CINCH 64
#define NCH   4
#define HH    192
#define WW    192
#define HWSZ  (HH*WW)        // 36864
#define NPIX  (BB*HWSZ)      // 294912
#define NBKD  (NPIX/256)     // 1152
#define NHD   (BB*HH)        // 1536
#define BIGF  384.0f         // h + w

typedef _Float16 v16h __attribute__((ext_vector_type(16)));
typedef float    v8f  __attribute__((ext_vector_type(8)));

// ---------------------------------------------------------------------------
// Kernel 1: projection GEMM via WMMA.
// logits_s[p, n] = sum_c feat[p, c] * W[n, c] + b[n], stored pixel-major
// (p*4 + n). One wave computes a 16-pixel x 16-col tile (cols 4..15 padded 0).
// Two v_wmma_f32_16x16x32_f16 cover K = 64.
// ---------------------------------------------------------------------------
__global__ __launch_bounds__(256) void proj_wmma(
    const float* __restrict__ feat, const float* __restrict__ Wp,
    const float* __restrict__ bp, float* __restrict__ logits)
{
  const int lane = threadIdx.x & 31;
  const int wid  = blockIdx.x * (blockDim.x >> 5) + (threadIdx.x >> 5);
  const int P    = wid * 16;                 // base pixel of tile (HWSZ%16==0)
  const int b    = P / HWSZ;
  const int hw   = P % HWSZ;
  const int n    = lane & 15;                // A row / B,D column
  const int hf   = lane >> 4;                // lane half selects K range

  const float* fbase = feat + (size_t)b * CINCH * HWSZ + hw + n;

  v16h A0, A1, B0, B1;
#pragma unroll
  for (int e = 0; e < 16; ++e) {
    // 16-bit A/B layout (ISA 7.12.2): pairs per VGPR, half-wave K offset
    const int K = ((e >> 3) << 4) + hf * 8 + (e & 7);   // 0..31
    A0[e] = (_Float16)fbase[K * HWSZ];
    A1[e] = (_Float16)fbase[(K + 32) * HWSZ];
    const int wn = (n < NCH) ? n : 0;
    float w0 = Wp[wn * CINCH + K];
    float w1 = Wp[wn * CINCH + K + 32];
    B0[e] = (_Float16)((n < NCH) ? w0 : 0.f);
    B1[e] = (_Float16)((n < NCH) ? w1 : 0.f);
  }
  const float bias = (n < NCH) ? bp[n] : 0.f;
  v8f acc;
#pragma unroll
  for (int i = 0; i < 8; ++i) acc[i] = bias;   // accumulator carries +b_proj

  acc = __builtin_amdgcn_wmma_f32_16x16x32_f16(false, A0, false, B0,
                                               (short)0, acc, false, false);
  acc = __builtin_amdgcn_wmma_f32_16x16x32_f16(false, A1, false, B1,
                                               (short)0, acc, false, false);
  if (n < NCH) {
#pragma unroll
    for (int i = 0; i < 8; ++i) {
      const int m = i + hf * 8;                // D layout: rows 0-7 / 8-15
      logits[(P + m) * NCH + n] = acc[i];
    }
  }
}

__device__ inline float blk_reduce256(float v, float* red) {
  const int tid = threadIdx.x;
  red[tid] = v; __syncthreads();
  for (int s = 128; s > 0; s >>= 1) {
    if (tid < s) red[tid] += red[tid + s];
    __syncthreads();
  }
  return red[0];
}

// ---------------------------------------------------------------------------
// Kernel 2: edge mask + boundary-KD KL, per-pixel, per-block partial sums.
// ---------------------------------------------------------------------------
__global__ __launch_bounds__(256) void bkd_kernel(
    const float* __restrict__ logits, const float* __restrict__ teacher,
    const float* __restrict__ label, float* __restrict__ partials)
{
  __shared__ float red[256];
  const int p  = blockIdx.x * 256 + threadIdx.x;   // NPIX == 1152*256 exactly
  const int b  = p / HWSZ, hw = p % HWSZ;
  const int h  = hw / WW, w = hw % WW;

  float s[3], t[3];
#pragma unroll
  for (int j = 0; j < 3; ++j) {
    const int k = j + 1;
    const float* L = label + ((size_t)b * NCH + k) * HWSZ;
    float sum9 = 0.f;
#pragma unroll
    for (int dh = -1; dh <= 1; ++dh)
#pragma unroll
      for (int dw = -1; dw <= 1; ++dw) {
        const int h2 = h + dh, w2 = w + dw;
        if (h2 >= 0 && h2 < HH && w2 >= 0 && w2 < WW) sum9 += L[h2 * WW + w2];
      }
    const float e = (sum9 > 0.f && sum9 < 9.f) ? 1.f : 0.f;
    s[j] = logits[p * NCH + k] * e * 0.25f;                       // /TAU
    t[j] = teacher[((size_t)b * NCH + k) * HWSZ + hw] * e * 0.25f;
  }
  const float ms = fmaxf(s[0], fmaxf(s[1], s[2]));
  const float lse = ms + logf(expf(s[0]-ms) + expf(s[1]-ms) + expf(s[2]-ms));
  const float mt = fmaxf(t[0], fmaxf(t[1], t[2]));
  float qe[3]; float qs = 0.f;
#pragma unroll
  for (int j = 0; j < 3; ++j) { qe[j] = expf(t[j] - mt); qs += qe[j]; }
  const float lqs = logf(qs);
  float acc = 0.f;
#pragma unroll
  for (int j = 0; j < 3; ++j) {
    const float q    = qe[j] / qs;
    const float logq = (t[j] - mt) - lqs;        // stable log(q), q>0 always
    const float pj   = s[j] - lse;
    acc += q * (logq - pj);                      // xlogy(q,q) - q*p
  }
  const float bsum = blk_reduce256(acc, red);
  if (threadIdx.x == 0) partials[blockIdx.x] = bsum;
}

// ---------------------------------------------------------------------------
// Kernel 3 (per class c): student prob mask (>0.5) + teacher argmax mask,
// plus per-image popcounts for the nontrivial test. One block per image.
// ---------------------------------------------------------------------------
__global__ __launch_bounds__(256) void mask_kernel(
    const float* __restrict__ logits, const float* __restrict__ teacher,
    unsigned char* __restrict__ maskP, unsigned char* __restrict__ maskT,
    int* __restrict__ counts, int c)
{
  __shared__ int redP[256], redT[256];
  const int b = blockIdx.x, tid = threadIdx.x;
  int cp = 0, ct = 0;
  for (int hw = tid; hw < HWSZ; hw += 256) {
    const int p = b * HWSZ + hw;
    const float l0 = logits[p*4+0], l1 = logits[p*4+1];
    const float l2 = logits[p*4+2], l3 = logits[p*4+3];
    const float m  = fmaxf(fmaxf(l0, l1), fmaxf(l2, l3));
    const float e0 = expf(l0-m), e1 = expf(l1-m), e2 = expf(l2-m), e3 = expf(l3-m);
    const float inv = 1.f / (e0 + e1 + e2 + e3);
    const float pc  = ((c == 0) ? e1 : (c == 1) ? e2 : e3) * inv;
    const int mp = pc > 0.5f;
    const float* T = teacher + (size_t)b * NCH * HWSZ + hw;
    const float t0 = T[0], t1 = T[HWSZ], t2 = T[2*HWSZ], t3 = T[3*HWSZ];
    int am = 0; float best = t0;                  // first-max tiebreak (jnp)
    if (t1 > best) { best = t1; am = 1; }
    if (t2 > best) { best = t2; am = 2; }
    if (t3 > best) { best = t3; am = 3; }
    const int mt = (am == c + 1);
    maskP[p] = (unsigned char)mp; maskT[p] = (unsigned char)mt;
    cp += mp; ct += mt;
  }
  redP[tid] = cp; redT[tid] = ct; __syncthreads();
  for (int s = 128; s > 0; s >>= 1) {
    if (tid < s) { redP[tid] += redP[tid+s]; redT[tid] += redT[tid+s]; }
    __syncthreads();
  }
  if (tid == 0) { counts[b] = redP[0]; counts[8 + b] = redT[0]; }
}

// ---------------------------------------------------------------------------
// Kernel 4: 1D row pass of the exact EDT (squared). One thread per (b,row),
// 4 fields at once (mask / ~mask for student and teacher). Two sweeps; the
// left-distance is staged in the destination buffer (same-thread store->load
// to same address is kept in order per ISA).
// ---------------------------------------------------------------------------
__global__ __launch_bounds__(256) void edt_rows(
    const unsigned char* __restrict__ maskP, const unsigned char* __restrict__ maskT,
    float* __restrict__ f2P, float* __restrict__ f2Pc,
    float* __restrict__ f2T, float* __restrict__ f2Tc)
{
  const int idx = blockIdx.x * 256 + threadIdx.x;
  if (idx >= BB * HH) return;
  const int b = idx / HH, r = idx % HH;
  const int base = b * HWSZ + r * WW;
  const unsigned char* mp = maskP + base;
  const unsigned char* mt = maskT + base;

  float lP = -BIGF, lPc = -BIGF, lT = -BIGF, lTc = -BIGF;
  for (int w = 0; w < WW; ++w) {
    const int vP = mp[w], vT = mt[w];
    const float fw = (float)w;
    if (!vP) lP = fw;  if (vP) lPc = fw;
    if (!vT) lT = fw;  if (vT) lTc = fw;
    f2P [base+w] = fw - lP;   f2Pc[base+w] = fw - lPc;
    f2T [base+w] = fw - lT;   f2Tc[base+w] = fw - lTc;
  }
  float rP = 2.f*BIGF, rPc = 2.f*BIGF, rT = 2.f*BIGF, rTc = 2.f*BIGF;
  for (int w = WW - 1; w >= 0; --w) {
    const int vP = mp[w], vT = mt[w];
    const float fw = (float)w;
    if (!vP) rP = fw;  if (vP) rPc = fw;
    if (!vT) rT = fw;  if (vT) rTc = fw;
    float fP  = fminf(fminf(f2P [base+w], rP  - fw), BIGF);
    float fPc = fminf(fminf(f2Pc[base+w], rPc - fw), BIGF);
    float fT  = fminf(fminf(f2T [base+w], rT  - fw), BIGF);
    float fTc = fminf(fminf(f2Tc[base+w], rTc - fw), BIGF);
    f2P [base+w] = fP  * fP;   f2Pc[base+w] = fPc * fPc;
    f2T [base+w] = fT  * fT;   f2Tc[base+w] = fTc * fTc;
  }
}

// ---------------------------------------------------------------------------
// Kernel 5: column lower-envelope d2[r,c] = min_r' f2[r',c] + (r-r')^2,
// fused with the Hausdorff error accumulation. Block = (b, out-row);
// threads 0..191 = columns; f2 fields are L2-resident (4.7 MB).
// pdt^ALPHA (ALPHA=2) == d2(m) + d2(~m) since one term is always zero.
// ---------------------------------------------------------------------------
__global__ __launch_bounds__(256) void hd_col_kernel(
    const float* __restrict__ f2P, const float* __restrict__ f2Pc,
    const float* __restrict__ f2T, const float* __restrict__ f2Tc,
    const float* __restrict__ logits, const unsigned char* __restrict__ maskT,
    const int* __restrict__ counts, int c, float* __restrict__ partials)
{
  __shared__ float red[256];
  const int tid = threadIdx.x;
  const int b = blockIdx.x / HH, r = blockIdx.x % HH;
  float acc = 0.f;
  if (tid < WW) {
    const int w = tid;
    const int cbase = b * HWSZ + w;
    float d2p = 1e30f, d2pc = 1e30f, d2t = 1e30f, d2tc = 1e30f;
    for (int r2 = 0; r2 < HH; ++r2) {
      const float dr  = (float)(r - r2);
      const float dr2 = dr * dr;
      const int o = cbase + r2 * WW;
      d2p  = fminf(d2p,  f2P [o] + dr2);
      d2pc = fminf(d2pc, f2Pc[o] + dr2);
      d2t  = fminf(d2t,  f2T [o] + dr2);
      d2tc = fminf(d2tc, f2Tc[o] + dr2);
    }
    const int cntP = counts[b], cntT = counts[8 + b];
    const float pdt2 = (cntP > 0 && cntP < HWSZ) ? (d2p + d2pc) : 0.f;
    const float tdt2 = (cntT > 0 && cntT < HWSZ) ? (d2t + d2tc) : 0.f;
    const int p = b * HWSZ + r * WW + w;
    const float l0 = logits[p*4+0], l1 = logits[p*4+1];
    const float l2 = logits[p*4+2], l3 = logits[p*4+3];
    const float m  = fmaxf(fmaxf(l0, l1), fmaxf(l2, l3));
    const float e0 = expf(l0-m), e1 = expf(l1-m), e2 = expf(l2-m), e3 = expf(l3-m);
    const float pf = (((c == 0) ? e1 : (c == 1) ? e2 : e3)) / (e0+e1+e2+e3);
    const float tf = (float)maskT[p];
    const float d  = pf - tf;
    acc = d * d * (pdt2 + tdt2);
  }
  const float bsum = blk_reduce256(acc, red);
  if (tid == 0) partials[blockIdx.x] = bsum;
}

// ---------------------------------------------------------------------------
// Kernel 6: deterministic final reduction + loss constants -> d_out[0..1].
// ---------------------------------------------------------------------------
__global__ __launch_bounds__(256) void final_kernel(
    const float* __restrict__ bkd_part, const float* __restrict__ hd_part,
    float* __restrict__ out)
{
  __shared__ float red[256];
  const int tid = threadIdx.x;
  float s = 0.f;
  for (int i = tid; i < NBKD; i += 256) s += bkd_part[i];
  const float bkd_sum = blk_reduce256(s, red);
  __syncthreads();
  float hd[3];
  for (int c = 0; c < 3; ++c) {
    float a = 0.f;
    for (int i = tid; i < NHD; i += 256) a += hd_part[c * NHD + i];
    hd[c] = blk_reduce256(a, red);
    __syncthreads();
  }
  if (tid == 0) {
    const float kl  = bkd_sum / (float)BB;
    const float bkd = kl * 16.0f;                           // * TAU^2
    out[0] = bkd * 1.0f * 0.5f;                             // OVERALL_W*LOSS_W
    const float mean = (hd[0] + hd[1] + hd[2]) / (3.0f * (float)NPIX);
    out[1] = logf(mean + 1.0f) * 1.0f * 0.5f;               // OVERALL_W*(1-LOSS_W)
  }
}

// ---------------------------------------------------------------------------
extern "C" void kernel_launch(void* const* d_in, const int* in_sizes, int n_in,
                              void* d_out, int out_size, void* d_ws, size_t ws_size,
                              hipStream_t stream) {
  (void)in_sizes; (void)n_in; (void)out_size; (void)ws_size;
  const float* feat    = (const float*)d_in[0];
  const float* teacher = (const float*)d_in[1];
  const float* label   = (const float*)d_in[2];
  const float* Wp      = (const float*)d_in[3];
  const float* bp      = (const float*)d_in[4];
  float* out = (float*)d_out;

  char* ws = (char*)d_ws;
  size_t off = 0;
  auto carve = [&](size_t bytes) -> char* {
    char* p = ws + off;
    off = (off + bytes + 255) & ~(size_t)255;
    return p;
  };
  float*         logits   = (float*)carve(sizeof(float) * (size_t)NPIX * NCH);
  unsigned char* maskP    = (unsigned char*)carve(NPIX);
  unsigned char* maskT    = (unsigned char*)carve(NPIX);
  int*           counts   = (int*)carve(sizeof(int) * 16);
  float*         f2P      = (float*)carve(sizeof(float) * NPIX);
  float*         f2Pc     = (float*)carve(sizeof(float) * NPIX);
  float*         f2T      = (float*)carve(sizeof(float) * NPIX);
  float*         f2Tc     = (float*)carve(sizeof(float) * NPIX);
  float*         bkd_part = (float*)carve(sizeof(float) * NBKD);
  float*         hd_part  = (float*)carve(sizeof(float) * 3 * NHD);

  // 1) WMMA projection: 18432 16-pixel tiles, 8 waves/block -> 2304 blocks
  proj_wmma<<<NPIX / 16 / 8, 256, 0, stream>>>(feat, Wp, bp, logits);
  // 2) boundary-KD partial sums
  bkd_kernel<<<NBKD, 256, 0, stream>>>(logits, teacher, label, bkd_part);
  // 3) Hausdorff term per foreground class (buffers reused sequentially)
  for (int c = 0; c < 3; ++c) {
    mask_kernel<<<BB, 256, 0, stream>>>(logits, teacher, maskP, maskT, counts, c);
    edt_rows<<<(BB * HH + 255) / 256, 256, 0, stream>>>(maskP, maskT,
                                                        f2P, f2Pc, f2T, f2Tc);
    hd_col_kernel<<<BB * HH, 256, 0, stream>>>(f2P, f2Pc, f2T, f2Tc, logits,
                                               maskT, counts, c, hd_part + c * NHD);
  }
  // 4) deterministic final reduction -> (bkd_loss, hd_loss)
  final_kernel<<<1, 256, 0, stream>>>(bkd_part, hd_part, out);
}
